// MultiHeadAttention_52862457480066
// MI455X (gfx1250) — matrix operand order repaired and checked by reading, not compiled, not run on hardware
//
#include <hip/hip_runtime.h>
#include <hip/hip_bf16.h>
#include <stdint.h>

// ---------------------------------------------------------------------------
// Problem constants
// ---------------------------------------------------------------------------
#define BATCH 2
#define SEQ   1024
#define DMODEL 512
#define HEADS 16
#define DHEAD 32
#define QKVC  (3 * HEADS * DHEAD)   // 1536
#define ROWS  (BATCH * SEQ)         // 2048
#define NEG_INF_F (-1e9f)
#define LN_EPS 1e-5f

typedef _Float16 v16h __attribute__((ext_vector_type(16)));
typedef float    v8f  __attribute__((ext_vector_type(8)));
typedef unsigned int v4u __attribute__((ext_vector_type(4)));
typedef float    v2f  __attribute__((ext_vector_type(2)));
typedef v4u __attribute__((may_alias)) v4u_a;
typedef v2f __attribute__((may_alias)) v2f_a;

// ---------------------------------------------------------------------------
// WMMA helpers (v_wmma_f32_16x16x32_f16, wave32)
// ---------------------------------------------------------------------------
__device__ __forceinline__ v8f wmma16x16x32(v16h a, v16h b, v8f c) {
    return __builtin_amdgcn_wmma_f32_16x16x32_f16(
        /*neg_a=*/false, a, /*neg_b=*/false, b,
        /*c_mod=*/(short)0, c, /*reuse_a=*/false, /*reuse_b=*/false);
}

// A fragment: 16x32 f16, row-major source X[ld], rows m0..m0+15, cols k0..k0+31.
// Lane L: M = L%16; halves 0..7 -> K = k0 + 8*(L/16) + i; halves 8..15 -> +16.
__device__ __forceinline__ v16h frag_a(const _Float16* __restrict__ X, int ld,
                                       int m0, int k0, int lane) {
    int m  = m0 + (lane & 15);
    int kb = k0 + ((lane >> 4) << 3);
    const _Float16* p = X + (size_t)m * ld + kb;
    union { v16h v; v4u u[2]; } t;
    t.u[0] = *(const v4u_a*)(p);
    t.u[1] = *(const v4u_a*)(p + 16);
    return t.v;
}

// B fragment where B[k][n] = W[n][k] (W row-major, ld): 32x16.
// Lane L: N = L%16; halves 0..7 -> K = k0 + 8*(L/16) + i; halves 8..15 -> +16.
__device__ __forceinline__ v16h frag_bt(const _Float16* __restrict__ W, int ld,
                                        int n0, int k0, int lane) {
    int n  = n0 + (lane & 15);
    int kb = k0 + ((lane >> 4) << 3);
    const _Float16* p = W + (size_t)n * ld + kb;
    union { v16h v; v4u u[2]; } t;
    t.u[0] = *(const v4u_a*)(p);
    t.u[1] = *(const v4u_a*)(p + 16);
    return t.v;
}

// A fragment from a 16x32 f16 LDS tile (row stride 32 halves).
__device__ __forceinline__ v16h frag_a_lds(const _Float16* P, int lane) {
    int m  = lane & 15;
    int kb = (lane >> 4) << 3;
    const _Float16* p = P + m * 32 + kb;
    union { v16h v; v4u u[2]; } t;
    t.u[0] = *(const v4u_a*)(p);
    t.u[1] = *(const v4u_a*)(p + 16);
    return t.v;
}

// ---------------------------------------------------------------------------
// Kernel 1: fp32 -> fp16 conversion of x, att_w, ff_w
// ---------------------------------------------------------------------------
#define XH_N (ROWS * DMODEL)        // 1048576
#define AW_N (QKVC * DMODEL)        //  786432
#define FW_N (DMODEL * DMODEL)      //  262144

__global__ void __launch_bounds__(256)
k_convert(const float* __restrict__ x, const float* __restrict__ aw,
          const float* __restrict__ fw,
          _Float16* __restrict__ xh, _Float16* __restrict__ awh,
          _Float16* __restrict__ fwh) {
    int i = blockIdx.x * blockDim.x + threadIdx.x;
    const int total = XH_N + AW_N + FW_N;
    const int stride = gridDim.x * blockDim.x;
    for (; i < total; i += stride) {
        if (i < XH_N)                 xh[i] = (_Float16)x[i];
        else if (i < XH_N + AW_N)     awh[i - XH_N] = (_Float16)aw[i - XH_N];
        else                          fwh[i - XH_N - AW_N] = (_Float16)fw[i - XH_N - AW_N];
    }
}

// ---------------------------------------------------------------------------
// Kernel 2: QKV GEMM  out[m,c] = sum_k xh[m,k]*awh[c,k]
// M=2048, N=1536, K=512.  Block = 4 waves; wave computes 16x64.
// Stores: qh/kh as [b*H+h][n][dh] f16 (q pre-scaled 1/sqrt(32)),
//         vth as [b*H+h][dh][n] f16 (transposed for PV B-fragments).
// ---------------------------------------------------------------------------
__global__ void __launch_bounds__(128)
k_qkv(const _Float16* __restrict__ xh, const _Float16* __restrict__ awh,
      _Float16* __restrict__ qh, _Float16* __restrict__ kh,
      _Float16* __restrict__ vth) {
    const int lane = threadIdx.x & 31;
    const int wave = threadIdx.x >> 5;
    const int m0 = blockIdx.x * 16;
    const int c0 = blockIdx.y * 256 + wave * 64;

    v8f acc[4] = {v8f{}, v8f{}, v8f{}, v8f{}};
    for (int kk = 0; kk < DMODEL; kk += 32) {
        v16h a = frag_a(xh, DMODEL, m0, kk, lane);
#pragma unroll
        for (int nt = 0; nt < 4; ++nt) {
            v16h b = frag_bt(awh, DMODEL, c0 + nt * 16, kk, lane);
            acc[nt] = wmma16x16x32(a, b, acc[nt]);
        }
    }

    const float qscale = 0.17677669529663687f;  // 1/sqrt(32)
#pragma unroll
    for (int nt = 0; nt < 4; ++nt) {
        int c = c0 + nt * 16 + (lane & 15);
        int which = c >> 9;      // 0=q 1=k 2=v
        int rem   = c & 511;
        int h = rem >> 5;
        int d = rem & 31;
#pragma unroll
        for (int r = 0; r < 8; ++r) {
            int m = m0 + r + ((lane >> 4) << 3);
            int b = m >> 10;
            int n = m & 1023;
            int head = b * HEADS + h;
            float v = acc[nt][r];
            if (which == 0)
                qh[((size_t)head * SEQ + n) * DHEAD + d] = (_Float16)(v * qscale);
            else if (which == 1)
                kh[((size_t)head * SEQ + n) * DHEAD + d] = (_Float16)v;
            else
                vth[((size_t)head * DHEAD + d) * SEQ + n] = (_Float16)v;
        }
    }
}

// ---------------------------------------------------------------------------
// Kernel 3: flash-style attention per (i-tile, head, batch). 1 wave / block.
// ---------------------------------------------------------------------------
__global__ void __launch_bounds__(32)
k_attn(const _Float16* __restrict__ qh, const _Float16* __restrict__ kh,
       const _Float16* __restrict__ vth,
       const float* __restrict__ pdist, const float* __restrict__ angle,
       const float* __restrict__ adj, const unsigned char* __restrict__ mask,
       const float* __restrict__ gamma_p, const float* __restrict__ gamma_adj,
       const float* __restrict__ w_bias, _Float16* __restrict__ att_h) {
    __shared__ __align__(32) _Float16 Pt[16 * 32];

    const int lane = threadIdx.x & 31;
    const int i0   = blockIdx.x * 16;
    const int h    = blockIdx.y;
    const int b    = blockIdx.z;
    const int head = b * HEADS + h;

    const _Float16* Q  = qh  + (size_t)head * SEQ * DHEAD;
    const _Float16* Km = kh  + (size_t)head * SEQ * DHEAD;
    const _Float16* Vt = vth + (size_t)head * DHEAD * SEQ;
    const float* PD = pdist + (size_t)b * SEQ * SEQ;
    const float* AJ = adj   + (size_t)b * SEQ * SEQ;
    const float* AN = angle + (size_t)b * SEQ * SEQ * 2;
    const unsigned char* MK = mask + (size_t)b * SEQ;

    const float gp  = gamma_p[h];
    const float ga  = gamma_adj[h];
    const float wb0 = w_bias[h * 2 + 0];
    const float wb1 = w_bias[h * 2 + 1];

    const v16h aq = frag_a(Q, DHEAD, i0, 0, lane);

    float mrow[8], lrow[8];
    v8f o0 = v8f{}, o1 = v8f{};
#pragma unroll
    for (int r = 0; r < 8; ++r) { mrow[r] = -3.0e38f; lrow[r] = 0.f; }

    for (int j0 = 0; j0 < SEQ; j0 += 32) {
        float sv[2][8];
#pragma unroll
        for (int sub = 0; sub < 2; ++sub) {
            const int jb = j0 + sub * 16;
            v16h bk = frag_bt(Km, DHEAD, jb, 0, lane);
            v8f s = wmma16x16x32(aq, bk, v8f{});
            const int j = jb + (lane & 15);
            const bool msk = MK[j] != 0;
#pragma unroll
            for (int r = 0; r < 8; ++r) {
                int i = i0 + r + ((lane >> 4) << 3);
                size_t off = (size_t)i * SEQ + j;
                float sc = s[r] - gp * PD[off] + ga * AJ[off];
                v2f an = *(const v2f_a*)(AN + 2 * off);
                sc += an.x * wb0 + an.y * wb1;
                sv[sub][r] = msk ? NEG_INF_F : sc;
            }
        }
        // online softmax across both 16-col subtiles of this 32-col block
#pragma unroll
        for (int r = 0; r < 8; ++r) {
            float mt = fmaxf(sv[0][r], sv[1][r]);
            mt = fmaxf(mt, __shfl_xor(mt, 1, 32));
            mt = fmaxf(mt, __shfl_xor(mt, 2, 32));
            mt = fmaxf(mt, __shfl_xor(mt, 4, 32));
            mt = fmaxf(mt, __shfl_xor(mt, 8, 32));
            float mnew  = fmaxf(mrow[r], mt);
            float scale = __expf(mrow[r] - mnew);
            mrow[r] = mnew;
            lrow[r] *= scale;
            o0[r] *= scale;
            o1[r] *= scale;
            float p0 = __expf(sv[0][r] - mnew);
            float p1 = __expf(sv[1][r] - mnew);
            float t = p0 + p1;
            t += __shfl_xor(t, 1, 32);
            t += __shfl_xor(t, 2, 32);
            t += __shfl_xor(t, 4, 32);
            t += __shfl_xor(t, 8, 32);
            lrow[r] += t;
            int row = r + ((lane >> 4) << 3);
            Pt[row * 32 + 0 * 16 + (lane & 15)] = (_Float16)p0;
            Pt[row * 32 + 1 * 16 + (lane & 15)] = (_Float16)p1;
        }
        __syncthreads();
        v16h ap  = frag_a_lds(Pt, lane);
        v16h bv0 = frag_bt(Vt, SEQ, 0,  j0, lane);   // d-tile 0 (d=0..15)
        v16h bv1 = frag_bt(Vt, SEQ, 16, j0, lane);   // d-tile 1 (d=16..31)
        o0 = wmma16x16x32(ap, bv0, o0);
        o1 = wmma16x16x32(ap, bv1, o1);
        __syncthreads();
    }

#pragma unroll
    for (int r = 0; r < 8; ++r) {
        int i = i0 + r + ((lane >> 4) << 3);
        float inv = 1.f / lrow[r];
        size_t rowoff = ((size_t)(b * SEQ + i)) * DMODEL + h * DHEAD;
        att_h[rowoff + (lane & 15)]      = (_Float16)(o0[r] * inv);
        att_h[rowoff + 16 + (lane & 15)] = (_Float16)(o1[r] * inv);
    }
}

// ---------------------------------------------------------------------------
// Kernel 4: FF GEMM + residual + LayerNorm.  Block = 8 waves = full 512 cols
// of one 16-row stripe. y -> LDS -> per-row LN.
// ---------------------------------------------------------------------------
__global__ void __launch_bounds__(256)
k_ffln(const _Float16* __restrict__ att_h, const _Float16* __restrict__ fwh,
       const float* __restrict__ x, const float* __restrict__ ff_b,
       const float* __restrict__ ln_w, const float* __restrict__ ln_b,
       float* __restrict__ out) {
    __shared__ float Yt[16 * DMODEL];   // 32 KB

    const int lane = threadIdx.x & 31;
    const int wave = threadIdx.x >> 5;
    const int m0 = blockIdx.x * 16;
    const int c0 = wave * 64;

    v8f acc[4] = {v8f{}, v8f{}, v8f{}, v8f{}};
    for (int kk = 0; kk < DMODEL; kk += 32) {
        v16h a = frag_a(att_h, DMODEL, m0, kk, lane);
#pragma unroll
        for (int nt = 0; nt < 4; ++nt) {
            v16h b = frag_bt(fwh, DMODEL, c0 + nt * 16, kk, lane);
            acc[nt] = wmma16x16x32(a, b, acc[nt]);
        }
    }
#pragma unroll
    for (int nt = 0; nt < 4; ++nt) {
        int c = c0 + nt * 16 + (lane & 15);
#pragma unroll
        for (int r = 0; r < 8; ++r) {
            int row = r + ((lane >> 4) << 3);
            int m = m0 + row;
            float y = acc[nt][r] + x[(size_t)m * DMODEL + c] + ff_b[c];
            Yt[row * DMODEL + c] = y;
        }
    }
    __syncthreads();

#pragma unroll
    for (int rr = 0; rr < 2; ++rr) {
        int row = wave * 2 + rr;
        int m = m0 + row;
        float s = 0.f, s2 = 0.f;
        for (int c = lane; c < DMODEL; c += 32) {
            float y = Yt[row * DMODEL + c];
            s += y;
            s2 += y * y;
        }
#pragma unroll
        for (int off = 1; off < 32; off <<= 1) {
            s  += __shfl_xor(s,  off, 32);
            s2 += __shfl_xor(s2, off, 32);
        }
        float mu  = s  * (1.f / DMODEL);
        float var = s2 * (1.f / DMODEL) - mu * mu;
        float rs  = rsqrtf(var + LN_EPS);
        for (int c = lane; c < DMODEL; c += 32) {
            float y = Yt[row * DMODEL + c];
            out[(size_t)m * DMODEL + c] = (y - mu) * rs * ln_w[c] + ln_b[c];
        }
    }
}

// ---------------------------------------------------------------------------
// Launch
// ---------------------------------------------------------------------------
extern "C" void kernel_launch(void* const* d_in, const int* in_sizes, int n_in,
                              void* d_out, int out_size, void* d_ws, size_t ws_size,
                              hipStream_t stream) {
    const float* x        = (const float*)d_in[0];
    const float* pdist    = (const float*)d_in[1];
    const float* angle    = (const float*)d_in[2];
    const float* adj      = (const float*)d_in[3];
    const unsigned char* mask = (const unsigned char*)d_in[4];
    const float* gamma_p  = (const float*)d_in[5];
    const float* gamma_adj= (const float*)d_in[6];
    const float* w_bias   = (const float*)d_in[7];
    const float* att_w    = (const float*)d_in[8];
    const float* ff_w     = (const float*)d_in[9];
    const float* ff_b     = (const float*)d_in[10];
    const float* ln_w     = (const float*)d_in[11];
    const float* ln_b     = (const float*)d_in[12];

    // workspace layout (bytes)
    const size_t XH_B  = (size_t)XH_N * 2;                 // 2,097,152
    const size_t AW_B  = (size_t)AW_N * 2;                 // 1,572,864
    const size_t FW_B  = (size_t)FW_N * 2;                 //   524,288
    const size_t HD_B  = (size_t)BATCH * HEADS * SEQ * DHEAD * 2; // 2,097,152
    size_t off = 0;
    char* ws = (char*)d_ws;
    _Float16* xh   = (_Float16*)(ws + off); off += XH_B;
    _Float16* awh  = (_Float16*)(ws + off); off += AW_B;
    _Float16* fwh  = (_Float16*)(ws + off); off += FW_B;
    _Float16* qh   = (_Float16*)(ws + off); off += HD_B;
    _Float16* kh   = (_Float16*)(ws + off); off += HD_B;
    _Float16* vth  = (_Float16*)(ws + off); off += HD_B;
    _Float16* atth = (_Float16*)(ws + off); off += XH_B;
    if (off > ws_size) return;  // workspace too small; bail deterministically

    k_convert<<<2048, 256, 0, stream>>>(x, att_w, ff_w, xh, awh, fwh);
    k_qkv<<<dim3(ROWS / 16, QKVC / 256), 128, 0, stream>>>(xh, awh, qh, kh, vth);
    k_attn<<<dim3(SEQ / 16, HEADS, BATCH), 32, 0, stream>>>(
        qh, kh, vth, pdist, angle, adj, mask, gamma_p, gamma_adj, w_bias, atth);
    k_ffln<<<ROWS / 16, 256, 0, stream>>>(atth, fwh, x, ff_b, ln_w, ln_b,
                                          (float*)d_out);
}
